// PerformerModel_66709432041472
// MI455X (gfx1250) — matrix operand order repaired and checked by reading, not compiled, not run on hardware
//
#include <hip/hip_runtime.h>
#include <cstdint>
#include <cstddef>

typedef __attribute__((ext_vector_type(16))) _Float16 v16h;
typedef __attribute__((ext_vector_type(8)))  float    v8f;

// ---------------- problem constants ----------------
#define BB      2
#define TT      8
#define HHW     48
#define SPAT    (HHW*HHW)          // 2304
#define SS      (TT*SPAT)          // 18432 tokens per batch
#define DD      64
#define NHEADS  4
#define DHD     16
#define MFEAT   32
#define FFD     256
#define NTOK    (BB*SS)            // 36864
#define NCHAIN  (BB*NHEADS)        // 8
#define CHUNK   144
#define NCHUNK  128                // 144*128 == 18432
#define STATE   544                // 32*16 ctx + 32 ksum

#define DN_     0.5f               // 16^-0.25
#define RATIO_  0.17677669529663687f // 32^-0.5
#define FEPS_   1e-4f

__device__ __forceinline__ float softplus_f(float x) {
    return (x > 20.f) ? x : log1pf(expf(x));
}
__device__ __forceinline__ float gelu_f(float x) {
    return 0.5f * x * (1.f + erff(x * 0.70710678118654752f));
}

// ---------------- preproc: 1x1x1 conv + softplus -> tokens (raw reshape) ----
__global__ void preproc_kernel(const float* __restrict__ x,
                               const float* __restrict__ pw,
                               const float* __restrict__ pb,
                               float* __restrict__ tok) {
    int idx = blockIdx.x * blockDim.x + threadIdx.x;
    const int per = SS * DD;                      // 1179648 per batch
    if (idx >= BB * per) return;
    int b = idx / per, i = idx - b * per;
    int d = i / SS;                               // channel in [B,D,T,H,W] view
    int sp = i - d * SS;                          // t*2304 + r
    int t = sp / SPAT, r = sp - t * SPAT;
    float acc = pb[d];
    #pragma unroll
    for (int c = 0; c < 3; ++c) {
        float xv = x[((size_t)(b * 3 + c) * TT + t) * SPAT + r];
        acc += xv * pw[d * 3 + c];
    }
    tok[idx] = softplus_f(acc);
}

// ---------------- layernorm over last dim 64, emit f16 ----------------------
__global__ void layernorm_kernel(const float* __restrict__ in,
                                 const float* __restrict__ g,
                                 const float* __restrict__ bta,
                                 _Float16* __restrict__ out) {
    int t = blockIdx.x * blockDim.x + threadIdx.x;
    if (t >= NTOK) return;
    const float* row = in + (size_t)t * DD;
    float s = 0.f, s2 = 0.f;
    for (int c = 0; c < DD; ++c) { float v = row[c]; s += v; s2 += v * v; }
    float mu = s * (1.f / DD);
    float var = s2 * (1.f / DD) - mu * mu;
    float rstd = rsqrtf(var + 1e-5f);
    for (int c = 0; c < DD; ++c)
        out[(size_t)t * DD + c] = (_Float16)((row[c] - mu) * rstd * g[c] + bta[c]);
}

// ---------------- WMMA GEMM: [NTOK x K](f16) @ [K x N](f32) -----------------
// One wave per 16x16 tile. The K x 16 f32 weight panel is staged into LDS via
// the gfx1250 async copy path (global_load_async_to_lds_b128 + s_wait_asynccnt),
// then B fragments are assembled from LDS. K is compile-time so the k-loop
// fully unrolls into K/32 v_wmma_f32_16x16x32_f16 ops.
// flags: bit0 = accumulate into outF32, bit1 = exact GELU
template<int K>
__global__ void gemm_wmma_kernel(const _Float16* __restrict__ A,
                                 const float* __restrict__ W,
                                 const float* __restrict__ bias,
                                 float* __restrict__ outF32,
                                 _Float16* __restrict__ outF16,
                                 int N, int flags) {
    __shared__ float wpanel[K * 16];          // K rows x 16 cols, f32
    const int lane    = threadIdx.x & 31;
    const int halfSel = lane >> 4;
    const int mr      = lane & 15;
    const int tileM   = blockIdx.x;
    const int tileN   = blockIdx.y;
    const int col     = tileN * 16 + mr;

    // ---- async copy W panel (K*64 bytes) into LDS; 512B per instruction ----
    {
        const unsigned ldsBase = (unsigned)(uintptr_t)wpanel; // wave-relative LDS addr (low 32b of flat)
        const int chunk = lane & 3;           // four 16B chunks per 64B row
        const int rbase = lane >> 2;          // 8 rows per instruction
        #pragma unroll
        for (int i = 0; i < K / 8; ++i) {
            const int row = i * 8 + rbase;
            unsigned lds = ldsBase + (unsigned)(row * 64 + chunk * 16);
            uint64_t ga  = (uint64_t)(uintptr_t)(W + (size_t)row * N + tileN * 16 + chunk * 4);
            asm volatile("global_load_async_to_lds_b128 %0, %1, off"
                         :: "v"(lds), "v"(ga) : "memory");
        }
        asm volatile("s_wait_asynccnt 0" ::: "memory");
    }

    const _Float16* arow = A + (size_t)(tileM * 16 + mr) * K;
    __builtin_prefetch(arow + K, 0, 0);       // global_prefetch_b8 (next A row)

    v8f c = {};
    #pragma unroll
    for (int ks = 0; ks < K / 32; ++ks) {
        const int k0 = ks * 32;
        v16h a, b;
        #pragma unroll
        for (int j = 0; j < 8; ++j) {
            // A 16x32 f16 layout: lanes0-15 K{0..7,16..23}, lanes16-31 K{8..15,24..31}
            int kbA = ((j & 4) << 2) + (halfSel << 3) + ((j & 3) << 1);
            a[2 * j]     = arow[k0 + kbA];
            a[2 * j + 1] = arow[k0 + kbA + 1];
            // B 32x16 f16 layout: lanes0-15 K{0..7,16..23}, lanes16-31 K{8..15,24..31}
            int kbB = ((j & 4) << 2) + (halfSel << 3) + ((j & 3) << 1);
            b[2 * j]     = (_Float16)wpanel[(k0 + kbB) * 16 + mr];
            b[2 * j + 1] = (_Float16)wpanel[(k0 + kbB + 1) * 16 + mr];
        }
        c = __builtin_amdgcn_wmma_f32_16x16x32_f16(false, a, false, b,
                                                   (short)0, c, false, false);
    }
    const float bcol = bias ? bias[col] : 0.f;
    #pragma unroll
    for (int r = 0; r < 8; ++r) {
        int row = tileM * 16 + r + (halfSel << 3);
        float v = c[r] + bcol;
        if (flags & 2) v = gelu_f(v);
        size_t o = (size_t)row * N + col;
        if (outF32) { if (flags & 1) outF32[o] += v; else outF32[o] = v; }
        if (outF16) outF16[o] = (_Float16)v;
    }
}

// ---------------- FAVOR+ query features (per-row max) -----------------------
__global__ void featq_kernel(const float* __restrict__ q,
                             const float* __restrict__ proj,
                             float* __restrict__ qp) {
    int idx = blockIdx.x * blockDim.x + threadIdx.x;  // chain*S + s
    if (idx >= NCHAIN * SS) return;
    int chain = idx / SS, s = idx - chain * SS;
    int b = chain >> 2, h = chain & 3;
    const float* qrow = q + ((size_t)(b * SS + s)) * DD + h * DHD;
    float qd[DHD]; float diag = 0.f;
    #pragma unroll
    for (int i = 0; i < DHD; ++i) { qd[i] = qrow[i] * DN_; diag += 0.5f * qd[i] * qd[i]; }
    float dash[MFEAT]; float mx = -3.4e38f;
    #pragma unroll
    for (int j = 0; j < MFEAT; ++j) {
        float acc = 0.f;
        #pragma unroll
        for (int i = 0; i < DHD; ++i) acc += qd[i] * proj[j * DHD + i];
        dash[j] = acc; mx = fmaxf(mx, acc);
    }
    float* out = qp + (size_t)idx * MFEAT;
    #pragma unroll
    for (int j = 0; j < MFEAT; ++j)
        out[j] = RATIO_ * (expf(dash[j] - diag - mx) + FEPS_);
}

// ---------------- FAVOR+ key features pass 1: dash/diag + block max ---------
__global__ void featk1_kernel(const float* __restrict__ k,
                              const float* __restrict__ proj,
                              float* __restrict__ dashk,
                              float* __restrict__ diagk,
                              float* __restrict__ blockmax) {
    __shared__ float sm[256];
    int idx = blockIdx.x * blockDim.x + threadIdx.x;
    float rowmax = -3.4e38f;
    if (idx < NCHAIN * SS) {
        int chain = idx / SS, s = idx - chain * SS;
        int b = chain >> 2, h = chain & 3;
        const float* krow = k + ((size_t)(b * SS + s)) * DD + h * DHD;
        float kd[DHD]; float diag = 0.f;
        #pragma unroll
        for (int i = 0; i < DHD; ++i) { kd[i] = krow[i] * DN_; diag += 0.5f * kd[i] * kd[i]; }
        float* out = dashk + (size_t)idx * MFEAT;
        #pragma unroll
        for (int j = 0; j < MFEAT; ++j) {
            float acc = 0.f;
            #pragma unroll
            for (int i = 0; i < DHD; ++i) acc += kd[i] * proj[j * DHD + i];
            out[j] = acc; rowmax = fmaxf(rowmax, acc);
        }
        diagk[idx] = diag;
    }
    sm[threadIdx.x] = rowmax;
    __syncthreads();
    for (int st = 128; st > 0; st >>= 1) {
        if (threadIdx.x < st) sm[threadIdx.x] = fmaxf(sm[threadIdx.x], sm[threadIdx.x + st]);
        __syncthreads();
    }
    if (threadIdx.x == 0) blockmax[blockIdx.x] = sm[0];
}

__global__ void maxreduce_kernel(const float* __restrict__ blockmax, int n,
                                 float* __restrict__ gmax) {
    __shared__ float sm[256];
    float m = -3.4e38f;
    for (int i = threadIdx.x; i < n; i += 256) m = fmaxf(m, blockmax[i]);
    sm[threadIdx.x] = m;
    __syncthreads();
    for (int st = 128; st > 0; st >>= 1) {
        if (threadIdx.x < st) sm[threadIdx.x] = fmaxf(sm[threadIdx.x], sm[threadIdx.x + st]);
        __syncthreads();
    }
    if (threadIdx.x == 0) gmax[0] = sm[0];
}

// ---------------- key features pass 2: apply global max, in place -----------
__global__ void featk2_kernel(float* __restrict__ dashk,
                              const float* __restrict__ diagk,
                              const float* __restrict__ gmax) {
    size_t idx = (size_t)blockIdx.x * blockDim.x + threadIdx.x;
    if (idx >= (size_t)NCHAIN * SS * MFEAT) return;
    float mx = gmax[0];
    dashk[idx] = RATIO_ * (expf(dashk[idx] - diagk[idx >> 5] - mx) + FEPS_);
}

// ---------------- causal linear attention: chunked scan ---------------------
// lane d (0..15) owns column d of ctx[32][16]; state per chunk = ctx + ksum
__global__ void scanA_kernel(const float* __restrict__ kp,
                             const float* __restrict__ v,
                             float* __restrict__ cs) {
    const int lane = threadIdx.x & 31;
    if (lane >= 16) return;
    const int chunk = blockIdx.x, chain = blockIdx.y;
    const int b = chain >> 2, h = chain & 3, d = lane;
    float ctx[MFEAT], ksum[MFEAT];
    #pragma unroll
    for (int m = 0; m < MFEAT; ++m) { ctx[m] = 0.f; ksum[m] = 0.f; }
    const int s0 = chunk * CHUNK;
    for (int si = 0; si < CHUNK; ++si) {
        const int s = s0 + si;
        const float* krow = kp + ((size_t)chain * SS + s) * MFEAT;
        const float vd = v[((size_t)(b * SS + s)) * DD + h * DHD + d];
        #pragma unroll
        for (int m = 0; m < MFEAT; ++m) {
            float km = krow[m];
            ksum[m] += km;
            ctx[m]  += km * vd;
        }
    }
    float* base = cs + ((size_t)chain * NCHUNK + chunk) * STATE;
    #pragma unroll
    for (int m = 0; m < MFEAT; ++m) base[m * 16 + d] = ctx[m];
    if (d == 0) {
        #pragma unroll
        for (int m = 0; m < MFEAT; ++m) base[512 + m] = ksum[m];
    }
}

__global__ void scanB_kernel(float* __restrict__ cs) {
    const int chain = blockIdx.x;
    const int e = threadIdx.x;            // 0..543
    float run = 0.f;
    for (int c = 0; c < NCHUNK; ++c) {
        size_t o = ((size_t)chain * NCHUNK + c) * STATE + e;
        float t = cs[o];
        cs[o] = run;
        run += t;
    }
}

__global__ void scanC_kernel(const float* __restrict__ qp,
                             const float* __restrict__ kp,
                             const float* __restrict__ v,
                             const float* __restrict__ cs,
                             _Float16* __restrict__ attn) {
    const int lane = threadIdx.x & 31;
    if (lane >= 16) return;
    const int chunk = blockIdx.x, chain = blockIdx.y;
    const int b = chain >> 2, h = chain & 3, d = lane;
    const float* base = cs + ((size_t)chain * NCHUNK + chunk) * STATE;
    float ctx[MFEAT], ksum[MFEAT];
    #pragma unroll
    for (int m = 0; m < MFEAT; ++m) { ctx[m] = base[m * 16 + d]; ksum[m] = base[512 + m]; }
    const int s0 = chunk * CHUNK;
    for (int si = 0; si < CHUNK; ++si) {
        const int s = s0 + si;
        const float* krow = kp + ((size_t)chain * SS + s) * MFEAT;
        const float* qrow = qp + ((size_t)chain * SS + s) * MFEAT;
        const float vd = v[((size_t)(b * SS + s)) * DD + h * DHD + d];
        float num = 0.f, den = 0.f;
        #pragma unroll
        for (int m = 0; m < MFEAT; ++m) {
            float km = krow[m], qm = qrow[m];
            ksum[m] += km;
            ctx[m]  += km * vd;
            den += qm * ksum[m];
            num += qm * ctx[m];
        }
        attn[((size_t)(b * SS + s)) * DD + h * DHD + d] = (_Float16)(num / den);
    }
}

// ---------------- readout: mean over T, concat x[:,2,0] ---------------------
__global__ void pool_kernel(const float* __restrict__ tok,
                            const float* __restrict__ x,
                            float* __restrict__ pooled) {
    int idx = blockIdx.x * blockDim.x + threadIdx.x;
    if (idx >= BB * 65 * SPAT) return;
    int b = idx / (65 * SPAT), rem = idx - b * 65 * SPAT;
    int ch = rem / SPAT, p = rem - ch * SPAT;
    float val;
    if (ch < DD) {
        float s = 0.f;
        #pragma unroll
        for (int t = 0; t < TT; ++t)
            s += tok[(size_t)b * SS * DD + (size_t)ch * SS + t * SPAT + p];
        val = s * (1.f / TT);
    } else {
        val = x[((size_t)(b * 3 + 2) * TT + 0) * SPAT + p];   // x[:,2,0]
    }
    pooled[idx] = val;
}

__global__ void conv_kernel(const float* __restrict__ pooled,
                            const float* __restrict__ w,
                            const float* __restrict__ cb,
                            float* __restrict__ cout) {
    int idx = blockIdx.x * blockDim.x + threadIdx.x;
    if (idx >= BB * SPAT) return;
    int b = idx / SPAT, p = idx - b * SPAT;
    int y = p / HHW, xx = p - y * HHW;
    float acc = cb[0];
    for (int ci = 0; ci < 65; ++ci) {
        const float* pc = pooled + ((size_t)b * 65 + ci) * SPAT;
        #pragma unroll
        for (int ky = 0; ky < 5; ++ky) {
            int iy = y + ky - 2;
            if ((unsigned)iy >= (unsigned)HHW) continue;
            #pragma unroll
            for (int kx = 0; kx < 5; ++kx) {
                int ix = xx + kx - 2;
                if ((unsigned)ix >= (unsigned)HHW) continue;
                acc += pc[iy * HHW + ix] * w[ci * 25 + ky * 5 + kx];
            }
        }
    }
    cout[idx] = acc;
}

__global__ void final_kernel(const float* __restrict__ cout,
                             const float* __restrict__ rdw,
                             const float* __restrict__ rdb,
                             float* __restrict__ out) {
    __shared__ float sm[256];
    int b = blockIdx.x;
    float s = 0.f;
    for (int i = threadIdx.x; i < SPAT; i += 256) s += cout[b * SPAT + i];
    sm[threadIdx.x] = s;
    __syncthreads();
    for (int st = 128; st > 0; st >>= 1) {
        if (threadIdx.x < st) sm[threadIdx.x] += sm[threadIdx.x + st];
        __syncthreads();
    }
    if (threadIdx.x == 0) out[b] = (sm[0] * (1.f / SPAT)) * rdw[0] + rdb[0];
}

// ---------------- launcher --------------------------------------------------
extern "C" void kernel_launch(void* const* d_in, const int* in_sizes, int n_in,
                              void* d_out, int out_size, void* d_ws, size_t ws_size,
                              hipStream_t stream) {
    (void)in_sizes; (void)n_in; (void)out_size; (void)ws_size;
    const float* x     = (const float*)d_in[0];
    const float* pre_w = (const float*)d_in[1];
    const float* pre_b = (const float*)d_in[2];
    const float* ln1_g = (const float*)d_in[3];
    const float* ln1_b = (const float*)d_in[4];
    const float* wq    = (const float*)d_in[5];
    const float* wk    = (const float*)d_in[6];
    const float* wv    = (const float*)d_in[7];
    const float* wo    = (const float*)d_in[8];
    const float* bo    = (const float*)d_in[9];
    const float* proj  = (const float*)d_in[10];
    const float* ln2_g = (const float*)d_in[11];
    const float* ln2_b = (const float*)d_in[12];
    const float* ff_w1 = (const float*)d_in[13];
    const float* ff_b1 = (const float*)d_in[14];
    const float* ff_w2 = (const float*)d_in[15];
    const float* ff_b2 = (const float*)d_in[16];
    const float* tgt_w = (const float*)d_in[17];
    const float* tgt_b = (const float*)d_in[18];
    const float* rd_w  = (const float*)d_in[19];
    const float* rd_b  = (const float*)d_in[20];
    float* out = (float*)d_out;

    // ---- workspace layout (bytes) ----
    char* w = (char*)d_ws;
    constexpr size_t SZ_TOK  = (size_t)NTOK * DD * 4;
    constexpr size_t SZ_LN   = (size_t)NTOK * DD * 2;
    constexpr size_t SZ_QKV  = (size_t)NTOK * DD * 4;
    constexpr size_t SZ_FEAT = (size_t)NCHAIN * SS * MFEAT * 4;
    constexpr size_t SZ_DIAG = (size_t)NCHAIN * SS * 4;
    constexpr size_t SZ_RED  = 4096;
    constexpr size_t SZ_CS   = (size_t)NCHAIN * NCHUNK * STATE * 4;
    constexpr size_t SZ_ATTN = (size_t)NTOK * DD * 2;
    constexpr size_t SZ_FF1  = (size_t)NTOK * FFD * 2;
    constexpr size_t SZ_POOL = (size_t)BB * 65 * SPAT * 4;
    constexpr size_t SZ_CONV = (size_t)BB * SPAT * 4;

    size_t off = 0;
    float*    tok   = (float*)(w + off);    off += SZ_TOK;
    _Float16* lnb   = (_Float16*)(w + off); off += SZ_LN;
    float*    qbuf  = (float*)(w + off);    off += SZ_QKV;
    float*    kbuf  = (float*)(w + off);    off += SZ_QKV;
    float*    vbuf  = (float*)(w + off);    off += SZ_QKV;
    float*    qp    = (float*)(w + off);    off += SZ_FEAT;
    float*    kp    = (float*)(w + off);    off += SZ_FEAT;   // dashk -> kp in place
    float*    diagk = (float*)(w + off);    off += SZ_DIAG;
    float*    red   = (float*)(w + off);    off += SZ_RED;    // [0]=gmax, [1..]=blockmax
    float*    cs    = (float*)(w + off);    off += SZ_CS;
    _Float16* attn  = (_Float16*)(w + off); off += SZ_ATTN;
    _Float16* ff1   = (_Float16*)(w + off); off += SZ_FF1;
    float*    pooled= (float*)(w + off);    off += SZ_POOL;
    float*    convo = (float*)(w + off);    off += SZ_CONV;

    const dim3 blk256(256);
    const dim3 gemmBlk(32);

    // 1) preproc -> tokens
    preproc_kernel<<<dim3((BB * SS * DD) / 256), blk256, 0, stream>>>(x, pre_w, pre_b, tok);
    // 2) LN1 -> f16
    layernorm_kernel<<<dim3(NTOK / 256), blk256, 0, stream>>>(tok, ln1_g, ln1_b, lnb);
    // 3) q,k,v GEMMs (WMMA f16 -> f32)
    gemm_wmma_kernel<DD><<<dim3(NTOK / 16, DD / 16), gemmBlk, 0, stream>>>(lnb, wq, nullptr, qbuf, nullptr, DD, 0);
    gemm_wmma_kernel<DD><<<dim3(NTOK / 16, DD / 16), gemmBlk, 0, stream>>>(lnb, wk, nullptr, kbuf, nullptr, DD, 0);
    gemm_wmma_kernel<DD><<<dim3(NTOK / 16, DD / 16), gemmBlk, 0, stream>>>(lnb, wv, nullptr, vbuf, nullptr, DD, 0);
    // 4) feature maps
    featq_kernel<<<dim3((NCHAIN * SS) / 256), blk256, 0, stream>>>(qbuf, proj, qp);
    featk1_kernel<<<dim3((NCHAIN * SS) / 256), blk256, 0, stream>>>(kbuf, proj, kp, diagk, red + 1);
    maxreduce_kernel<<<dim3(1), blk256, 0, stream>>>(red + 1, (NCHAIN * SS) / 256, red);
    featk2_kernel<<<dim3((NCHAIN * SS * MFEAT) / 256), blk256, 0, stream>>>(kp, diagk, red);
    // 5) causal linear attention scan
    scanA_kernel<<<dim3(NCHUNK, NCHAIN), gemmBlk, 0, stream>>>(kp, vbuf, cs);
    scanB_kernel<<<dim3(NCHAIN), dim3(STATE), 0, stream>>>(cs);
    scanC_kernel<<<dim3(NCHUNK, NCHAIN), gemmBlk, 0, stream>>>(qp, kp, vbuf, cs, attn);
    // 6) tokens += attn @ wo + bo
    gemm_wmma_kernel<DD><<<dim3(NTOK / 16, DD / 16), gemmBlk, 0, stream>>>(attn, wo, bo, tok, nullptr, DD, 1);
    // 7) LN2 -> f16 (reuse lnb)
    layernorm_kernel<<<dim3(NTOK / 256), blk256, 0, stream>>>(tok, ln2_g, ln2_b, lnb);
    // 8) ff1 = gelu(h2 @ ff_w1 + ff_b1) -> f16
    gemm_wmma_kernel<DD><<<dim3(NTOK / 16, FFD / 16), gemmBlk, 0, stream>>>(lnb, ff_w1, ff_b1, nullptr, ff1, FFD, 2);
    // 9) tokens += ff1 @ ff_w2 + ff_b2   (K=256 -> 8 unrolled WMMAs)
    gemm_wmma_kernel<FFD><<<dim3(NTOK / 16, DD / 16), gemmBlk, 0, stream>>>(ff1, ff_w2, ff_b2, tok, nullptr, DD, 1);
    // 10) readout
    pool_kernel<<<dim3((BB * 65 * SPAT + 255) / 256), blk256, 0, stream>>>(tok, x, pooled);
    conv_kernel<<<dim3((BB * SPAT + 255) / 256), blk256, 0, stream>>>(pooled, tgt_w, tgt_b, convo);
    final_kernel<<<dim3(BB), blk256, 0, stream>>>(convo, rd_w, rd_b, out);
}
